// DocREModel_85229331022362
// MI455X (gfx1250) — compile-verified, matrix operand order checked by reading, not run on hardware
//
#include <hip/hip_runtime.h>
#include <math.h>

// ---- problem constants ----
constexpr int kNB = 4, kH = 12, kC = 1024, kHID = 768, kEMB = 512;
constexpr int kE = 22, kM = 4, kL = 16, kSPAN = 32, kTD = 20, kINTER = 256;
constexpr int kNN = kE + kE * kM + kL;       // 126
constexpr int kNREL = 3;
constexpr int kD0 = kEMB + kTD;              // 532
constexpr int kD0P = 544;                    // padded K (mult of 32)
constexpr int kGW = 22;                      // spatial width
constexpr int kPIX = kGW * kGW;              // 484
constexpr int kPIXP = 496;                   // padded pixels (mult of 16)
constexpr int kROWS = kNB * kNN;             // 504
constexpr int kROWSP = 512;                  // padded rows
constexpr int kPGW = 26;                     // padded spatial (2-halo)
constexpr int kPPIX = kPGW * kPGW;           // 676

typedef __attribute__((ext_vector_type(16))) __bf16 v16bf;
typedef __attribute__((ext_vector_type(8)))  float  v8f;

// K offset inside a 32-wide step for vgpr j (0..7), lane-half hl (0/1):
// lanes 0-15 hold K {0..7,16..23}; lanes 16-31 hold K {8..15,24..31}
__device__ __forceinline__ int koff(int j, int hl) {
    return ((j & 4) ? 16 : 0) + hl * 8 + (j & 3) * 2;
}

// contiguous-K fragment loader: merges into 4x global_load_b128
__device__ __forceinline__ v16bf load_ctg(const float* __restrict__ P, int hl) {
    v16bf a;
#pragma unroll
    for (int j = 0; j < 8; ++j) {
        float2 x = *(const float2*)(P + koff(j, hl));
        a[2 * j] = (__bf16)x.x;
        a[2 * j + 1] = (__bf16)x.y;
    }
    return a;
}

__device__ __forceinline__ v8f wmma_bf16(v16bf a, v16bf b, v8f c) {
    return __builtin_amdgcn_wmma_f32_16x16x32_bf16(false, a, false, b,
                                                   (short)0, c, false, false);
}

// --- generic batched GEMM: C = A * B, B given TRANSPOSED (N x K row-major) --
// 64(M)x16(N) per wave; one B fragment reused by 4 wmma; all loads contiguous.
__global__ void gemm_wmma(const float* __restrict__ A, long sA, int lda,
                          const float* __restrict__ Bt, long sB, int ldbT,
                          float* __restrict__ Cm, long sC, int ldc,
                          int N, int K,
                          const float* __restrict__ biasRow,
                          const float* __restrict__ biasCol,
                          const float* __restrict__ gRow,
                          const float* __restrict__ beRow, int relu, int storeT) {
    int lane = threadIdx.x & 31, wave = threadIdx.x >> 5;
    int hl = lane >> 4, r = lane & 15;
    int rowBase = blockIdx.x * 64;                 // 4 row tiles per wave
    int colBase = (blockIdx.y * 4 + wave) * 16;
    if (colBase >= N) return;                      // uniform per wave
    int col = colBase + r;
    const float* Ab = A + (size_t)blockIdx.z * sA + (size_t)(rowBase + r) * lda;
    const float* Bb = Bt + (size_t)blockIdx.z * sB + (size_t)col * ldbT;
    float* Cb = Cm + (size_t)blockIdx.z * sC;
    v8f acc[4] = {};
    for (int kk = 0; kk < K; kk += 32) {
        __builtin_prefetch(Ab + kk + 64, 0, 1);
        v16bf b  = load_ctg(Bb + kk, hl);
        v16bf a0 = load_ctg(Ab + kk, hl);
        v16bf a1 = load_ctg(Ab + (size_t)16 * lda + kk, hl);
        v16bf a2 = load_ctg(Ab + (size_t)32 * lda + kk, hl);
        v16bf a3 = load_ctg(Ab + (size_t)48 * lda + kk, hl);
        acc[0] = wmma_bf16(a0, b, acc[0]);
        acc[1] = wmma_bf16(a1, b, acc[1]);
        acc[2] = wmma_bf16(a2, b, acc[2]);
        acc[3] = wmma_bf16(a3, b, acc[3]);
    }
#pragma unroll
    for (int t = 0; t < 4; ++t) {
#pragma unroll
        for (int j = 0; j < 8; ++j) {
            int rr = rowBase + t * 16 + hl * 8 + j;
            float v = acc[t][j];
            if (biasRow) v += biasRow[rr];
            if (biasCol) v += biasCol[col];
            if (gRow) v = v * gRow[rr] + beRow[rr];
            if (relu) v = fmaxf(v, 0.0f);
            if (storeT) Cb[(size_t)col * ldc + rr] = v;       // N-major (contig per lane)
            else        Cb[(size_t)rr * ldc + col] = v;       // M-major
        }
    }
}

// ---- RGCN dense stage: Hout = relu(sum_r AH_r*Wrel_r + Hin*Wself) ----------
// Weights pre-transposed: (rel, N=512, K=d). All fragments contiguous.
__global__ void rgcn_wmma(const float* __restrict__ AH, const float* __restrict__ Hin,
                          const float* __restrict__ WrelT, const float* __restrict__ WselfT,
                          float* __restrict__ Hout, int d) {
    int lane = threadIdx.x & 31, wave = threadIdx.x >> 5;
    int hl = lane >> 4, r = lane & 15;
    int rowBase = blockIdx.x * 64;
    int colBase = (blockIdx.y * 4 + wave) * 16;
    int col = colBase + r;
    v8f acc[4] = {};
    for (int rel = 0; rel <= kNREL; ++rel) {
        const float* Ar = (rel < kNREL ? AH + (size_t)rel * kROWSP * d : Hin)
                          + (size_t)(rowBase + r) * d;
        const float* Br = (rel < kNREL ? WrelT + (size_t)rel * kEMB * d : WselfT)
                          + (size_t)col * d;
        for (int kk = 0; kk < d; kk += 32) {
            v16bf b  = load_ctg(Br + kk, hl);
            v16bf a0 = load_ctg(Ar + kk, hl);
            v16bf a1 = load_ctg(Ar + (size_t)16 * d + kk, hl);
            v16bf a2 = load_ctg(Ar + (size_t)32 * d + kk, hl);
            v16bf a3 = load_ctg(Ar + (size_t)48 * d + kk, hl);
            acc[0] = wmma_bf16(a0, b, acc[0]);
            acc[1] = wmma_bf16(a1, b, acc[1]);
            acc[2] = wmma_bf16(a2, b, acc[2]);
            acc[3] = wmma_bf16(a3, b, acc[3]);
        }
    }
#pragma unroll
    for (int t = 0; t < 4; ++t) {
#pragma unroll
        for (int j = 0; j < 8; ++j) {
            int rr = rowBase + t * 16 + hl * 8 + j;
            Hout[(size_t)rr * kEMB + col] = fmaxf(acc[t][j], 0.0f);
        }
    }
}

// ---- 5x5 SAME conv, implicit GEMM; zero-halo channel-last activations ------
// Wr: (Cout, 25, Cin) tap-major (contiguous); 16(out-ch) x 64(pixels) per wave.
__global__ void conv5x5_wmma(const float* __restrict__ Xp, const float* __restrict__ Wr,
                             const float* __restrict__ bias, float* __restrict__ Yp,
                             float* __restrict__ Ync, int Cin, int Cout) {
    int lane = threadIdx.x & 31, wave = threadIdx.x >> 5;
    int hl = lane >> 4, r = lane & 15;
    int rowBase = (blockIdx.x * 4 + wave) * 16;   // out-channel tile
    int pBase = blockIdx.y * 64;                  // 4 pixel tiles
    int n = blockIdx.z;
    int p[4], oh[4], ow[4];
#pragma unroll
    for (int t = 0; t < 4; ++t) {
        p[t] = pBase + t * 16 + r;
        int pc = p[t] < kPIX ? p[t] : kPIX - 1;   // clamp for loads
        oh[t] = pc / kGW;
        ow[t] = pc % kGW;
    }
    const float* Xn = Xp + (size_t)n * kPPIX * Cin;
    const float* Aw = Wr + (size_t)(rowBase + r) * (Cin * 25);
    v8f acc[4] = {};
#pragma unroll
    for (int kh = 0; kh < 5; ++kh) {
#pragma unroll
        for (int kw = 0; kw < 5; ++kw) {
            int sp = kh * 5 + kw;
            const float* Ap = Aw + (size_t)sp * Cin;
            const float* Bb0 = Xn + (size_t)((oh[0] + kh) * kPGW + (ow[0] + kw)) * Cin;
            const float* Bb1 = Xn + (size_t)((oh[1] + kh) * kPGW + (ow[1] + kw)) * Cin;
            const float* Bb2 = Xn + (size_t)((oh[2] + kh) * kPGW + (ow[2] + kw)) * Cin;
            const float* Bb3 = Xn + (size_t)((oh[3] + kh) * kPGW + (ow[3] + kw)) * Cin;
            for (int cc = 0; cc < Cin; cc += 32) {
                v16bf a  = load_ctg(Ap + cc, hl);
                v16bf b0 = load_ctg(Bb0 + cc, hl);
                v16bf b1 = load_ctg(Bb1 + cc, hl);
                v16bf b2 = load_ctg(Bb2 + cc, hl);
                v16bf b3 = load_ctg(Bb3 + cc, hl);
                acc[0] = wmma_bf16(a, b0, acc[0]);
                acc[1] = wmma_bf16(a, b1, acc[1]);
                acc[2] = wmma_bf16(a, b2, acc[2]);
                acc[3] = wmma_bf16(a, b3, acc[3]);
            }
        }
    }
#pragma unroll
    for (int t = 0; t < 4; ++t) {
        if (p[t] < kPIX) {
#pragma unroll
            for (int j = 0; j < 8; ++j) {
                int rr = rowBase + hl * 8 + j;
                float v = fmaxf(acc[t][j] + bias[rr], 0.0f);
                if (Ync) Ync[((size_t)n * Cout + rr) * kPIX + p[t]] = v;
                else Yp[((size_t)n * kPPIX + (size_t)(oh[t] + 2) * kPGW + (ow[t] + 2)) * Cout + rr] = v;
            }
        }
    }
}

// ---------------- layout-transform kernels ----------------
__global__ void zero_kernel(float* p, int n) {
    int g = blockIdx.x * blockDim.x + threadIdx.x;
    if (g < n) p[g] = 0.0f;
}

// dst (N x K) = transpose of src (K x N)
__global__ void transpose_kn(const float* __restrict__ src, float* __restrict__ dst,
                             int K, int N) {
    int gid = blockIdx.x * blockDim.x + threadIdx.x;
    if (gid >= K * N) return;
    int k = gid % K, nn = gid / K;
    dst[gid] = src[(size_t)k * N + nn];
}

// batch of square transposes: dst[b][n][k] = src[b][k][n]
__global__ void transpose_sq_batch(const float* __restrict__ src, float* __restrict__ dst,
                                   int nmat, int dim) {
    int gid = blockIdx.x * blockDim.x + threadIdx.x;
    if (gid >= nmat * dim * dim) return;
    int within = gid % (dim * dim);
    int b = gid / (dim * dim);
    int k = within % dim, nn = within / dim;
    dst[gid] = src[(size_t)b * dim * dim + (size_t)k * dim + nn];
}

// layer-0 RGCN weights: transpose + pad K 532->544 (zeros)
__global__ void w0t_kernel(const float* __restrict__ src, float* __restrict__ dst) {
    int gid = blockIdx.x * blockDim.x + threadIdx.x;
    if (gid >= kNREL * kEMB * kD0P) return;
    int k = gid % kD0P;
    int rest = gid / kD0P;
    int o = rest % kEMB, rel = rest / kEMB;
    dst[gid] = (k < kD0) ? src[((size_t)rel * kD0 + k) * kEMB + o] : 0.0f;
}

__global__ void ws0t_kernel(const float* __restrict__ src, float* __restrict__ dst) {
    int gid = blockIdx.x * blockDim.x + threadIdx.x;
    if (gid >= kEMB * kD0P) return;
    int k = gid % kD0P, o = gid / kD0P;
    dst[gid] = (k < kD0) ? src[(size_t)k * kEMB + o] : 0.0f;
}

// conv weights (Cout, Cin, 25) -> tap-major (Cout, 25, Cin)
__global__ void wconv_reorder(const float* __restrict__ src, float* __restrict__ dst,
                              int Cout, int Cin) {
    int gid = blockIdx.x * blockDim.x + threadIdx.x;
    if (gid >= Cout * 25 * Cin) return;
    int c = gid % Cin;
    int rest = gid / Cin;
    int sp = rest % 25, o = rest / 25;
    dst[gid] = src[((size_t)o * Cin + c) * 25 + sp];
}

// ---------------- model scalar kernels ----------------
__global__ void adj_set(float* adj) {
    int t = threadIdx.x;
    if (t >= kE * kM) return;
    int e = t / kM, m = t % kM;
    int mi = kE + e * kM + m;
    adj[0 * kNN * kNN + e * kNN + mi] = 1.0f;
    adj[0 * kNN * kNN + mi * kNN + e] = 1.0f;
    for (int m2 = 0; m2 < kM; ++m2)
        if (m2 != m) adj[1 * kNN * kNN + mi * kNN + (kE + e * kM + m2)] = 1.0f;
    int li = kE + kE * kM + (e * kM + m) % kL;
    adj[2 * kNN * kNN + mi * kNN + li] = 1.0f;
    adj[2 * kNN * kNN + li * kNN + mi] = 1.0f;
}

__global__ void adj_norm(float* adj) {
    __shared__ float sm[128];
    float* rowp = adj + (size_t)blockIdx.x * kNN;
    int t = threadIdx.x;
    float v = (t < kNN) ? rowp[t] : 0.0f;
    sm[t] = v;
    __syncthreads();
    for (int s = 64; s > 0; s >>= 1) {
        if (t < s) sm[t] += sm[t + s];
        __syncthreads();
    }
    float denom = sm[0] + 1e-5f;
    if (t < kNN) rowp[t] = v / denom;
}

// ea[n,e,c] = mean_{m,h} att[n,h,midx,c], normalized over c
__global__ void ea_kernel(const float* __restrict__ att, const int* __restrict__ midx,
                          float* __restrict__ ea) {
    __shared__ float vals[kC];
    __shared__ float red[256];
    int ne = blockIdx.x;
    int n = ne / kE, e = ne % kE;
    int t = threadIdx.x;
    float lsum = 0.0f;
    for (int c = t; c < kC; c += 256) {
        float s = 0.0f;
        for (int m = 0; m < kM; ++m) {
            int ci = midx[(n * kE + e) * kM + m];
            for (int h = 0; h < kH; ++h)
                s += att[(((size_t)n * kH + h) * kC + ci) * kC + c];
        }
        s *= (1.0f / (kM * kH));
        vals[c] = s;
        lsum += s;
    }
    red[t] = lsum;
    __syncthreads();
    for (int s2 = 128; s2 > 0; s2 >>= 1) {
        if (t < s2) red[t] += red[t + s2];
        __syncthreads();
    }
    float denom = red[0] + 1e-5f;
    for (int c = t; c < kC; c += 256)
        ea[((size_t)n * kE + e) * kC + c] = vals[c] / denom;
}

__global__ void erep_kernel(const float* __restrict__ seq, const int* __restrict__ midx,
                            float* __restrict__ erep) {
    int ne = blockIdx.x;
    int n = ne / kE, e = ne % kE;
    int t = threadIdx.x;
    int idx[kM];
    for (int m = 0; m < kM; ++m) idx[m] = midx[(n * kE + e) * kM + m];
    for (int d = t; d < kEMB; d += 256) {
        float x[kM], mx = -1e30f;
        for (int m = 0; m < kM; ++m) {
            x[m] = seq[((size_t)n * kC + idx[m]) * kEMB + d];
            mx = fmaxf(mx, x[m]);
        }
        float s = 0.0f;
        for (int m = 0; m < kM; ++m) s += expf(x[m] - mx);
        erep[((size_t)n * kE + e) * kEMB + d] = mx + logf(s);
    }
}

__global__ void link_kernel(const float* __restrict__ att, const float* __restrict__ seq,
                            const int* __restrict__ lstart, float* __restrict__ lrep) {
    __shared__ float part[kSPAN][8];
    __shared__ float colsum[kSPAN];
    int nl = blockIdx.x;
    int n = nl / kL, l = nl % kL;
    int i0 = lstart[n * kL + l];
    int t = threadIdx.x;
    int tcol = t >> 3, sub = t & 7;
    float s = 0.0f;
    for (int si = sub; si < kSPAN; si += 8) {
        for (int h = 0; h < kH; ++h)
            s += att[(((size_t)n * kH + h) * kC + (i0 + si)) * kC + (i0 + tcol)];
    }
    part[tcol][sub] = s;
    __syncthreads();
    if (t < kSPAN) {
        float cs = 0.0f;
        for (int q = 0; q < 8; ++q) cs += part[t][q];
        colsum[t] = cs * (1.0f / kH);
    }
    __syncthreads();
    for (int d = t; d < kEMB; d += 256) {
        float acc = 0.0f;
        for (int tt = 0; tt < kSPAN; ++tt)
            acc += colsum[tt] * seq[((size_t)n * kC + i0 + tt) * kEMB + d];
        lrep[((size_t)n * kL + l) * kEMB + d] = acc * (1.0f / kSPAN);
    }
}

// padded node matrix: (kROWSP, kD0P); pad rows/cols = 0
__global__ void h0_kernel(const float* __restrict__ erep, const float* __restrict__ seq,
                          const float* __restrict__ lrep, const float* __restrict__ temb,
                          const int* __restrict__ midx, float* __restrict__ h0) {
    int gid = blockIdx.x * blockDim.x + threadIdx.x;
    if (gid >= kROWSP * kD0P) return;
    int c = gid % kD0P;
    int rr = gid / kD0P;
    float v = 0.0f;
    if (rr < kROWS && c < kD0) {
        int n = rr / kNN, i = rr % kNN;
        if (c < kEMB) {
            if (i < kE) v = erep[((size_t)n * kE + i) * kEMB + c];
            else if (i < kE + kE * kM) {
                int ci = midx[n * kE * kM + (i - kE)];
                v = seq[((size_t)n * kC + ci) * kEMB + c];
            } else v = lrep[((size_t)n * kL + (i - kE - kE * kM)) * kEMB + c];
        } else {
            int ty = (i < kE) ? 0 : (i < kE + kE * kM) ? 1 : 2;
            v = temb[ty * kTD + (c - kEMB)];
        }
    }
    h0[gid] = v;
}

// AH[rel, n*NN+i, c] = sum_j adj[rel,i,j] * h[n*NN+j, c]   (padded stride d)
__global__ void ah_kernel(const float* __restrict__ adj, const float* __restrict__ h,
                          float* __restrict__ AH, int d) {
    int blk = blockIdx.x;
    int i = blk % kNN;
    int rest = blk / kNN;
    int n = rest % kNB, rel = rest / kNB;
    const float* arow = adj + ((size_t)rel * kNN + i) * kNN;
    const float* hb = h + (size_t)(n * kNN) * d;
    float* outp = AH + (size_t)rel * kROWSP * d + (size_t)(n * kNN + i) * d;
    for (int c = threadIdx.x; c < d; c += blockDim.x) {
        float acc = 0.0f;
        for (int j = 0; j < kNN; ++j) acc += arow[j] * hb[(size_t)j * d + c];
        outp[c] = acc;
    }
}

__global__ void ectx_kernel(const float* __restrict__ ea, const float* __restrict__ seq,
                            const float* __restrict__ hfin, float* __restrict__ ec) {
    int gid = blockIdx.x * blockDim.x + threadIdx.x;
    if (gid >= kNB * kE * kEMB) return;
    int dcol = gid % kEMB;
    int rest = gid / kEMB;
    int e = rest % kE, n = rest / kE;
    const float* ear = ea + ((size_t)n * kE + e) * kC;
    const float* sb = seq + (size_t)n * kC * kEMB + dcol;
    float acc = 0.0f;
    for (int c = 0; c < kC; ++c) acc += ear[c] * sb[(size_t)c * kEMB];
    ec[gid] = acc + hfin[(size_t)(n * kNN + e) * kEMB + dcol];
}

// fmap pixel-major: (NB, kPIXP, EMB); pad pixels zero
__global__ void fmap_kernel(const float* __restrict__ ec, float* __restrict__ fmap) {
    int gid = blockIdx.x * blockDim.x + threadIdx.x;
    if (gid >= kNB * kPIXP * kEMB) return;
    int c = gid % kEMB;
    int rest = gid / kEMB;
    int p = rest % kPIXP, n = rest / kPIXP;
    float v = 0.0f;
    if (p < kPIX) {
        int i = p / kGW, j = p % kGW;
        v = ec[((size_t)n * kE + i) * kEMB + c] * ec[((size_t)n * kE + j) * kEMB + c];
    }
    fmap[gid] = v;
}

__global__ void pool_kernel(const float* __restrict__ fmap, float* __restrict__ pooled) {
    int gid = blockIdx.x * blockDim.x + threadIdx.x;
    if (gid >= kNB * kEMB) return;
    int c = gid % kEMB, n = gid / kEMB;
    float s = 0.0f;
    for (int q = 0; q < kPIX; ++q)
        s += fmap[((size_t)n * kPIXP + q) * kEMB + c];
    pooled[gid] = s * (1.0f / kPIX);
}

__global__ void fc1_kernel(const float* __restrict__ pooled, const float* __restrict__ w1,
                           const float* __restrict__ b1, const float* __restrict__ g1,
                           const float* __restrict__ be1, float* __restrict__ tfc) {
    int gid = blockIdx.x * blockDim.x + threadIdx.x;
    if (gid >= kNB * kINTER) return;
    int o = gid % kINTER, n = gid / kINTER;
    float acc = 0.0f;
    for (int c = 0; c < kEMB; ++c) acc += w1[(size_t)o * kEMB + c] * pooled[n * kEMB + c];
    float v = g1[o] * (acc + b1[o]) + be1[o];
    tfc[gid] = fmaxf(v, 0.0f);
}

__global__ void fc2_kernel(const float* __restrict__ tfc, const float* __restrict__ w2,
                           const float* __restrict__ b2, const float* __restrict__ g2,
                           const float* __restrict__ be2, float* __restrict__ cb) {
    int gid = blockIdx.x * blockDim.x + threadIdx.x;
    if (gid >= kNB * kEMB) return;
    int c = gid % kEMB, n = gid / kEMB;
    float acc = 0.0f;
    for (int o = 0; o < kINTER; ++o) acc += w2[(size_t)c * kINTER + o] * tfc[n * kINTER + o];
    cb[gid] = g2[c] * (acc + b2[c]) + be2[c];
}

// fused gate -> channel-last zero-halo layout (NB, 26*26, EMB)
// fmap/sbuf are pixel-major (NB, kPIXP, EMB)
__global__ void fuse_kernel(const float* __restrict__ fmap, const float* __restrict__ sbuf,
                            const float* __restrict__ cb, float* __restrict__ fused) {
    int gid = blockIdx.x * blockDim.x + threadIdx.x;
    if (gid >= kNB * kPPIX * kEMB) return;
    int c = gid % kEMB;
    int rest = gid / kEMB;
    int pix = rest % kPPIX, n = rest / kPPIX;
    int ph = pix / kPGW, pw = pix % kPGW;
    float v = 0.0f;
    if (ph >= 2 && ph < 24 && pw >= 2 && pw < 24) {
        int p = (ph - 2) * kGW + (pw - 2);
        size_t o = ((size_t)n * kPIXP + p) * kEMB + c;
        float x = sbuf[o] + cb[n * kEMB + c];
        v = fmap[o] * (1.0f / (1.0f + expf(-x)));
    }
    fused[gid] = v;
}

// ---------------- host-side launch ----------------
extern "C" void kernel_launch(void* const* d_in, const int* in_sizes, int n_in,
                              void* d_out, int out_size, void* d_ws, size_t ws_size,
                              hipStream_t stream) {
    const float* seq_in  = (const float*)d_in[0];
    const float* att     = (const float*)d_in[1];
    const float* W_trans = (const float*)d_in[2];
    const float* b_trans = (const float*)d_in[3];
    const float* temb    = (const float*)d_in[4];
    const float* Wrel0   = (const float*)d_in[5];
    const float* Wself0  = (const float*)d_in[6];
    const float* Wrel    = (const float*)d_in[7];
    const float* Wself   = (const float*)d_in[8];
    const float* fs_w1 = (const float*)d_in[9];
    const float* fs_b1 = (const float*)d_in[10];
    const float* fs_g1 = (const float*)d_in[11];
    const float* fs_be1= (const float*)d_in[12];
    const float* fs_w2 = (const float*)d_in[13];
    const float* fs_b2 = (const float*)d_in[14];
    const float* fs_g2 = (const float*)d_in[15];
    const float* fs_be2= (const float*)d_in[16];
    const float* fc_w1 = (const float*)d_in[17];
    const float* fc_b1 = (const float*)d_in[18];
    const float* fc_g1 = (const float*)d_in[19];
    const float* fc_be1= (const float*)d_in[20];
    const float* fc_w2 = (const float*)d_in[21];
    const float* fc_b2 = (const float*)d_in[22];
    const float* fc_g2 = (const float*)d_in[23];
    const float* fc_be2= (const float*)d_in[24];
    const float* cr_w1 = (const float*)d_in[25];
    const float* cr_b1 = (const float*)d_in[26];
    const float* cr_w2 = (const float*)d_in[27];
    const float* cr_b2 = (const float*)d_in[28];
    const float* cr_w3 = (const float*)d_in[29];
    const float* cr_b3 = (const float*)d_in[30];
    const int* midx   = (const int*)d_in[31];
    const int* lstart = (const int*)d_in[32];
    float* out = (float*)d_out;
    (void)in_sizes; (void)n_in; (void)out_size; (void)ws_size;

    float* ws = (float*)d_ws;
    size_t off = 0;
    auto alloc = [&](size_t nelem) {
        float* p = ws + off;
        off += (nelem + 63) & ~(size_t)63;
        return p;
    };
    float* seq    = alloc((size_t)kNB * kC * kEMB);
    float* adj    = alloc((size_t)kNREL * kNN * kNN);
    float* ea     = alloc((size_t)kNB * kE * kC);
    float* erep   = alloc((size_t)kNB * kE * kEMB);
    float* lrep   = alloc((size_t)kNB * kL * kEMB);
    float* h0     = alloc((size_t)kROWSP * kD0P);
    float* AH     = alloc((size_t)kNREL * kROWSP * kD0P);
    float* hA     = alloc((size_t)kROWSP * kEMB);
    float* hB     = alloc((size_t)kROWSP * kEMB);
    float* WtT    = alloc((size_t)kEMB * kHID);           // W_trans^T (512x768)
    float* W0pT   = alloc((size_t)kNREL * kEMB * kD0P);   // layer-0 rel, T+pad
    float* Ws0pT  = alloc((size_t)kEMB * kD0P);           // layer-0 self, T+pad
    float* WrelT  = alloc((size_t)3 * kNREL * kEMB * kEMB);
    float* WselfT = alloc((size_t)3 * kEMB * kEMB);
    float* ec     = alloc((size_t)kNB * kE * kEMB);
    float* fmapb  = alloc((size_t)kNB * kPIXP * kEMB);    // pixel-major
    float* t1     = alloc((size_t)kNB * kPIXP * kINTER);  // pixel-major
    float* sbuf   = alloc((size_t)kNB * kPIXP * kEMB);    // pixel-major
    float* pooled = alloc((size_t)kNB * kEMB);
    float* tfc    = alloc((size_t)kNB * kINTER);
    float* cb     = alloc((size_t)kNB * kEMB);
    float* fusedp = alloc((size_t)kNB * kPPIX * kEMB);
    float* r1p    = alloc((size_t)kNB * kPPIX * kINTER);
    float* r2p    = alloc((size_t)kNB * kPPIX * kINTER);
    float* wr1    = alloc((size_t)kINTER * 25 * kEMB);    // conv1 weights tap-major
    float* wr2    = alloc((size_t)kINTER * 25 * kINTER);
    float* wr3    = alloc((size_t)kEMB * 25 * kINTER);

    // 0) weight layout transforms (tiny, memory-bound)
    int wtN = kEMB * kHID;
    transpose_kn<<<(wtN + 255) / 256, 256, 0, stream>>>(W_trans, WtT, kHID, kEMB);
    int w0N = kNREL * kEMB * kD0P;
    w0t_kernel<<<(w0N + 255) / 256, 256, 0, stream>>>(Wrel0, W0pT);
    int ws0N = kEMB * kD0P;
    ws0t_kernel<<<(ws0N + 255) / 256, 256, 0, stream>>>(Wself0, Ws0pT);
    int wrN = 3 * kNREL * kEMB * kEMB;
    transpose_sq_batch<<<(wrN + 255) / 256, 256, 0, stream>>>(Wrel, WrelT, 3 * kNREL, kEMB);
    int wsN = 3 * kEMB * kEMB;
    transpose_sq_batch<<<(wsN + 255) / 256, 256, 0, stream>>>(Wself, WselfT, 3, kEMB);
    int c1N = kINTER * 25 * kEMB;
    wconv_reorder<<<(c1N + 255) / 256, 256, 0, stream>>>(cr_w1, wr1, kINTER, kEMB);
    int c2N = kINTER * 25 * kINTER;
    wconv_reorder<<<(c2N + 255) / 256, 256, 0, stream>>>(cr_w2, wr2, kINTER, kINTER);
    int c3N = kEMB * 25 * kINTER;
    wconv_reorder<<<(c3N + 255) / 256, 256, 0, stream>>>(cr_w3, wr3, kEMB, kINTER);

    // 1) seq = X @ W_trans + b_trans  (4096x512x768) -- WMMA
    gemm_wmma<<<dim3(64, 8, 1), 128, 0, stream>>>(
        seq_in, 0L, kHID, WtT, 0L, kHID, seq, 0L, kEMB,
        kEMB, kHID, nullptr, b_trans, nullptr, nullptr, 0, 0);

    // 2) adjacency (device-generated constant)
    int adjN = kNREL * kNN * kNN;
    zero_kernel<<<(adjN + 255) / 256, 256, 0, stream>>>(adj, adjN);
    adj_set<<<1, 128, 0, stream>>>(adj);
    adj_norm<<<kNREL * kNN, 128, 0, stream>>>(adj);

    // 3) attention-derived features
    ea_kernel<<<kNB * kE, 256, 0, stream>>>(att, midx, ea);
    erep_kernel<<<kNB * kE, 256, 0, stream>>>(seq, midx, erep);
    link_kernel<<<kNB * kL, 256, 0, stream>>>(att, seq, lstart, lrep);

    // 4) padded node assembly
    int h0N = kROWSP * kD0P;
    h0_kernel<<<(h0N + 255) / 256, 256, 0, stream>>>(erep, seq, lrep, temb, midx, h0);

    // 5) RGCN (4 layers): SpMM (VALU) + dense WMMA, fully padded
    ah_kernel<<<kNREL * kNB * kNN, 256, 0, stream>>>(adj, h0, AH, kD0P);
    rgcn_wmma<<<dim3(8, 8, 1), 128, 0, stream>>>(AH, h0, W0pT, Ws0pT, hA, kD0P);

    ah_kernel<<<kNREL * kNB * kNN, 256, 0, stream>>>(adj, hA, AH, kEMB);
    rgcn_wmma<<<dim3(8, 8, 1), 128, 0, stream>>>(
        AH, hA, WrelT + 0L * kNREL * kEMB * kEMB, WselfT + 0L * kEMB * kEMB, hB, kEMB);

    ah_kernel<<<kNREL * kNB * kNN, 256, 0, stream>>>(adj, hB, AH, kEMB);
    rgcn_wmma<<<dim3(8, 8, 1), 128, 0, stream>>>(
        AH, hB, WrelT + 1L * kNREL * kEMB * kEMB, WselfT + 1L * kEMB * kEMB, hA, kEMB);

    ah_kernel<<<kNREL * kNB * kNN, 256, 0, stream>>>(adj, hA, AH, kEMB);
    rgcn_wmma<<<dim3(8, 8, 1), 128, 0, stream>>>(
        AH, hA, WrelT + 2L * kNREL * kEMB * kEMB, WselfT + 2L * kEMB * kEMB, hB, kEMB);

    // 6) entity context + bilinear map (pixel-major)
    int ecN = kNB * kE * kEMB;
    ectx_kernel<<<(ecN + 255) / 256, 256, 0, stream>>>(ea, seq, hB, ec);
    int fmN = kNB * kPIXP * kEMB;
    fmap_kernel<<<(fmN + 255) / 256, 256, 0, stream>>>(ec, fmapb);

    // 7) SE spatial path (two 1x1 convs + BN) -- WMMA, B pixel-major, C stored T
    gemm_wmma<<<dim3(4, 8, kNB), 128, 0, stream>>>(
        fs_w1, 0L, kEMB, fmapb, (long)kPIXP * kEMB, kEMB,
        t1, (long)kPIXP * kINTER, kINTER,
        kPIXP, kEMB, fs_b1, nullptr, fs_g1, fs_be1, 1, 1);
    gemm_wmma<<<dim3(8, 8, kNB), 128, 0, stream>>>(
        fs_w2, 0L, kINTER, t1, (long)kPIXP * kINTER, kINTER,
        sbuf, (long)kPIXP * kEMB, kEMB,
        kPIXP, kINTER, fs_b2, nullptr, fs_g2, fs_be2, 0, 1);

    // 8) SE channel path (tiny)
    pool_kernel<<<(kNB * kEMB + 255) / 256, 256, 0, stream>>>(fmapb, pooled);
    fc1_kernel<<<(kNB * kINTER + 255) / 256, 256, 0, stream>>>(pooled, fc_w1, fc_b1, fc_g1, fc_be1, tfc);
    fc2_kernel<<<(kNB * kEMB + 255) / 256, 256, 0, stream>>>(tfc, fc_w2, fc_b2, fc_g2, fc_be2, cb);

    // 9) sigmoid gate fuse into zero-halo channel-last layout
    int fuN = kNB * kPPIX * kEMB;
    fuse_kernel<<<(fuN + 255) / 256, 256, 0, stream>>>(fmapb, sbuf, cb, fusedp);

    // 10) zero halos of conv intermediates (border == SAME zero padding)
    int r1N = kNB * kPPIX * kINTER;
    zero_kernel<<<(r1N + 255) / 256, 256, 0, stream>>>(r1p, r1N);
    zero_kernel<<<(r1N + 255) / 256, 256, 0, stream>>>(r2p, r1N);

    // 11) three 5x5 SAME convs -- implicit-GEMM WMMA (dominant FLOPs)
    conv5x5_wmma<<<dim3(4, 8, kNB), 128, 0, stream>>>(fusedp, wr1, cr_b1, r1p, nullptr, kEMB, kINTER);
    conv5x5_wmma<<<dim3(4, 8, kNB), 128, 0, stream>>>(r1p, wr2, cr_b2, r2p, nullptr, kINTER, kINTER);
    conv5x5_wmma<<<dim3(8, 8, kNB), 128, 0, stream>>>(r2p, wr3, cr_b3, nullptr, out, kINTER, kEMB);
}
